// FreqSetAttention_23897198035720
// MI455X (gfx1250) — matrix operand order repaired and checked by reading, not compiled, not run on hardware
//
#include <hip/hip_runtime.h>

#define M_COMP    4
#define D_MODEL   128
#define N_HEADS   8
#define DQ        16
#define NUM_NODES 207
#define INPUT_LEN 12
#define BATCH     32
#define NPOS      (BATCH * NUM_NODES * INPUT_LEN)   // 79488 (divisible by 16)
#define POS_PER_B (NUM_NODES * INPUT_LEN)           // 2484
#define TILE      16

typedef __attribute__((ext_vector_type(16))) __bf16 v16bf;
typedef __attribute__((ext_vector_type(8)))  float  v8f;

union Frag16 { v16bf v; unsigned short u[16]; uint4 q[2]; };

__device__ __forceinline__ unsigned short f2bf(float f) {
    unsigned u = __float_as_uint(f);
    unsigned r = 0x7FFFu + ((u >> 16) & 1u);      // round-to-nearest-even
    return (unsigned short)((u + r) >> 16);
}

// ---------------------------------------------------------------------------
// Pre-pass: convert WQ/WK/WV (3 x 4 x 128 x 128 fp32) into bf16 tiles laid out
// as WMMA B-matrix fragments in this code's k order (32x16 per tile, per-lane 16 contiguous
// K values): lane = (k%32)/16*16 + e%16, slot = k%16. Weights then stream from
// L2 (192 MB) as two b128 loads per fragment per lane.
// ---------------------------------------------------------------------------
__global__ void prep_weights(const float* __restrict__ WQ,
                             const float* __restrict__ WK,
                             const float* __restrict__ WV,
                             unsigned short* __restrict__ wbf) {
    int idx = blockIdx.x * blockDim.x + threadIdx.x;   // 3*4*128*128 elements
    if (idx >= 3 * 4 * 128 * 128) return;
    int e    = idx & 127;
    int k    = (idx >> 7) & 127;
    int comp = (idx >> 14) & 3;
    int proj = idx >> 16;
    const float* W = (proj == 0) ? WQ : ((proj == 1) ? WK : WV);
    float w = W[((size_t)comp * 128 + k) * 128 + e];
    int kc = k >> 5, kk = k & 31, khalf = kk >> 4, kr = kk & 15;
    int ct = e >> 4, n = e & 15;
    int lane = khalf * 16 + n;
    size_t dst = ((((size_t)(proj * 4 + comp) * 4 + kc) * 8 + ct) * 32 + lane) * 16 + kr;
    wbf[dst] = f2bf(w);
}

// ---------------------------------------------------------------------------
// Fused kernel: one block = 16 positions x all 4 components x full pipeline.
// ---------------------------------------------------------------------------
__global__ void __launch_bounds__(256)
fsa_fused(const float* __restrict__ x,
          const float* __restrict__ bQ,
          const float* __restrict__ bK,
          const float* __restrict__ bV,
          const float* __restrict__ aw,
          const unsigned short* __restrict__ wbf,
          float* __restrict__ out) {
    extern __shared__ char smem[];
    unsigned short* xs   = (unsigned short*)smem;                       // 16*512 bf16 = 16 KB
    float*          qkvs = (float*)(smem + TILE * 512 * sizeof(unsigned short)); // 3*4*16*128 f32 = 96 KB
    float*          awexp = qkvs + 3 * 4 * TILE * 128;                  // 32 f32

    const int tid  = threadIdx.x;
    const int lane = tid & 31;          // wave32 lane id
    const int wave = tid >> 5;          // 8 waves
    const size_t tileBase = (size_t)blockIdx.x * (TILE * 512);

    // ---- Stage 1: load contiguous 32 KB x tile, convert to bf16 in LDS ----
    const float4* xv = (const float4*)(x + tileBase);
    #pragma unroll
    for (int it = 0; it < 8; ++it) {
        int vi = tid + it * 256;                      // 2048 float4 total
        float4 f = xv[vi];
        unsigned p0 = (unsigned)f2bf(f.x) | ((unsigned)f2bf(f.y) << 16);
        unsigned p1 = (unsigned)f2bf(f.z) | ((unsigned)f2bf(f.w) << 16);
        *(uint2*)(&xs[vi * 4]) = make_uint2(p0, p1);
    }
    if (tid < 32) awexp[tid] = expf(aw[tid]);         // exp(attn_weights[i][s])
    __syncthreads();

    // ---- Stage 2: Q/K/V projections with v_wmma_f32_16x16x32_bf16 ----
    const int comp = wave >> 1;                       // 2 waves per component
    const int half = wave & 1;

    // A fragments (16x32 bf16, k = 16*(lane/16) + slot, same order as B), shared across 12 jobs
    Frag16 afr[4];
    {
        const int m  = lane & 15;
        const int hi = lane >> 4;                     // 0: K 0-15; 1: K 16-31
        #pragma unroll
        for (int kc = 0; kc < 4; ++kc) {
            int base = m * 512 + comp * 128 + kc * 32 + hi * 16;
            afr[kc].q[0] = *(const uint4*)(&xs[base]);
            afr[kc].q[1] = *(const uint4*)(&xs[base + 8]);
        }
    }

    for (int jj = 0; jj < 12; ++jj) {
        int job  = half * 12 + jj;                    // 24 jobs = 3 proj x 8 col tiles
        int proj = job >> 3;
        int ct   = job & 7;
        const float* bp = (proj == 0) ? bQ : ((proj == 1) ? bK : bV);
        float bv = bp[comp * 128 + ct * 16 + (lane & 15)];
        v8f c = {bv, bv, bv, bv, bv, bv, bv, bv};     // C init = bias (broadcast over rows)

        #pragma unroll
        for (int kc = 0; kc < 4; ++kc) {
            size_t toff = ((((size_t)(proj * 4 + comp) * 4 + kc) * 8 + ct) * 32 + lane) * 16;
            Frag16 bfr;
            bfr.q[0] = *(const uint4*)(wbf + toff);
            bfr.q[1] = *(const uint4*)(wbf + toff + 8);
            c = __builtin_amdgcn_wmma_f32_16x16x32_bf16(
                    false, afr[kc].v, false, bfr.v, (short)0, c, false, false);
        }
        // D layout: VGPR r -> row m = r + (lane>=16)*8, col = ct*16 + lane%16
        const int m0  = (lane >> 4) * 8;
        const int col = ct * 16 + (lane & 15);
        float* dst = qkvs + ((proj * 4 + comp) * 16) * 128 + col;
        #pragma unroll
        for (int r = 0; r < 8; ++r) dst[(m0 + r) * 128] = c[r];
    }
    __syncthreads();

    // ---- Stage 3: per-(pos, comp, head) subset softmax + blend + residual ----
    #pragma unroll
    for (int rep = 0; rep < 2; ++rep) {
        int task = tid + rep * 256;                   // 512 = 16 pos * 4 comp * 8 heads
        int h = task & 7;
        int i = (task >> 3) & 3;
        int p = task >> 5;

        int others[3];
        { int cc = 0;
          #pragma unroll
          for (int j = 0; j < 4; ++j) if (j != i) others[cc++] = j; }

        float q[DQ], kk[3][DQ], vv[3][DQ];
        const float* qb = qkvs + ((0 * 4 + i) * 16 + p) * 128 + h * 16;
        #pragma unroll
        for (int d = 0; d < DQ; ++d) q[d] = qb[d];
        #pragma unroll
        for (int j = 0; j < 3; ++j) {
            const float* kb = qkvs + ((1 * 4 + others[j]) * 16 + p) * 128 + h * 16;
            const float* vb = qkvs + ((2 * 4 + others[j]) * 16 + p) * 128 + h * 16;
            #pragma unroll
            for (int d = 0; d < DQ; ++d) { kk[j][d] = kb[d]; vv[j][d] = vb[d]; }
        }

        float sc[3];
        #pragma unroll
        for (int j = 0; j < 3; ++j) {
            float s = 0.f;
            #pragma unroll
            for (int d = 0; d < DQ; ++d) s += q[d] * kk[j][d];
            sc[j] = s * 0.25f;                        // 1/sqrt(dq=16)
        }

        float acc[DQ];
        #pragma unroll
        for (int d = 0; d < DQ; ++d) acc[d] = 0.f;
        float sumw = 0.f;

        #pragma unroll
        for (int s = 1; s < 8; ++s) {
            float w = awexp[i * 8 + s];
            sumw += w;
            // MASKS[i][s]: bit j (MSB-first) selects others[j]
            const bool m0 = (s >> 2) & 1, m1 = (s >> 1) & 1, m2 = s & 1;
            float mx = -3.0e38f;
            if (m0) mx = fmaxf(mx, sc[0]);
            if (m1) mx = fmaxf(mx, sc[1]);
            if (m2) mx = fmaxf(mx, sc[2]);
            float e0 = m0 ? __expf(sc[0] - mx) : 0.f;
            float e1 = m1 ? __expf(sc[1] - mx) : 0.f;
            float e2 = m2 ? __expf(sc[2] - mx) : 0.f;
            float f = w / (e0 + e1 + e2);
            #pragma unroll
            for (int d = 0; d < DQ; ++d)
                acc[d] += f * (e0 * vv[0][d] + e1 * vv[1][d] + e2 * vv[2][d]);
        }
        float inv = 1.f / sumw;

        // Output permutation from reference reshape: dst row n = l*207 + v of the
        // source position; residual is x at the *destination* address.
        int P   = blockIdx.x * TILE + p;
        int b   = P / POS_PER_B;
        int rem = P - b * POS_PER_B;
        int v   = rem / INPUT_LEN;
        int l   = rem - v * INPUT_LEN;
        int n   = l * NUM_NODES + v;
        size_t dbase = ((size_t)b * POS_PER_B + n) * 512 + i * 128 + h * 16;

        const float4* xr = (const float4*)(x + dbase);
        float4*       ob = (float4*)(out + dbase);
        #pragma unroll
        for (int t = 0; t < 4; ++t) {
            float4 r = xr[t];
            float4 o;
            o.x = acc[t * 4 + 0] * inv + r.x;
            o.y = acc[t * 4 + 1] * inv + r.y;
            o.z = acc[t * 4 + 2] * inv + r.z;
            o.w = acc[t * 4 + 3] * inv + r.w;
            ob[t] = o;
        }
    }
}

extern "C" void kernel_launch(void* const* d_in, const int* in_sizes, int n_in,
                              void* d_out, int out_size, void* d_ws, size_t ws_size,
                              hipStream_t stream) {
    const float* x  = (const float*)d_in[0];
    const float* WQ = (const float*)d_in[1];
    const float* bQ = (const float*)d_in[2];
    const float* WK = (const float*)d_in[3];
    const float* bK = (const float*)d_in[4];
    const float* WV = (const float*)d_in[5];
    const float* bV = (const float*)d_in[6];
    const float* aw = (const float*)d_in[7];
    unsigned short* wbf = (unsigned short*)d_ws;   // 3*4*128*128 bf16 = 384 KB

    // weight bf16 repack (runs every call; deterministic, inputs untouched)
    prep_weights<<<(3 * 4 * 128 * 128 + 255) / 256, 256, 0, stream>>>(WQ, WK, WV, wbf);

    size_t lds = (size_t)TILE * 512 * sizeof(unsigned short)   // x tile bf16
               + (size_t)3 * 4 * TILE * 128 * sizeof(float)    // q/k/v f32
               + 32 * sizeof(float);                           // exp(attn_weights)
    fsa_fused<<<NPOS / TILE, 256, lds, stream>>>(x, bQ, bK, bV, aw, wbf, (float*)d_out);
}
